// RTFFSSM_model_24292335026643
// MI455X (gfx1250) — matrix operand order repaired and checked by reading, not compiled, not run on hardware
//
#include <hip/hip_runtime.h>
#include <hip/hip_fp16.h>
#include <math.h>

// ---------------------------------------------------------------------------
// Problem constants (match reference)
// ---------------------------------------------------------------------------
#define B_SZ   8
#define L_SEQ  2048
#define C_CH   128
#define H_ORD  64
#define NFFT   4096            // 2*L
#define D_PER_B (L_SEQ * C_CH) // 262144
#define TOT_ELEMS ((size_t)B_SZ * D_PER_B) // 2097152
#define M_ROWS (B_SZ * L_SEQ)  // 16384
#define N_ITER 5
#define AND_LAM 1e-6f

typedef _Float16 half16v __attribute__((ext_vector_type(16)));
typedef _Float16 half8v  __attribute__((ext_vector_type(8)));
typedef float    float8v __attribute__((ext_vector_type(8)));

// ---------------------------------------------------------------------------
// In-LDS radix-2 complex FFT, 4096 points, 256 threads
// ---------------------------------------------------------------------------
__device__ __forceinline__ void fft4096_lds(float* re, float* im, bool inverse) {
  const int t = threadIdx.x;
  for (int idx = t; idx < NFFT; idx += 256) {
    int r = (int)(__brev((unsigned)idx) >> 20);
    if (r > idx) {
      float tr = re[idx]; re[idx] = re[r]; re[r] = tr;
      float ti = im[idx]; im[idx] = im[r]; im[r] = ti;
    }
  }
  __syncthreads();
  const float sgn = inverse ? 1.0f : -1.0f;
  for (int s = 1; s <= 12; ++s) {
    const int half = 1 << (s - 1);
    const int m = half << 1;
    for (int bi = t; bi < (NFFT >> 1); bi += 256) {
      const int grp = bi >> (s - 1);
      const int pos = bi & (half - 1);
      const int i0 = grp * m + pos;
      const int i1 = i0 + half;
      float ang = sgn * 6.28318530717958647692f * (float)pos / (float)m;
      float wi, wr;
      __sincosf(ang, &wi, &wr);
      float xr = re[i1], xi = im[i1];
      float tr = wr * xr - wi * xi;
      float ti = wr * xi + wi * xr;
      float ur = re[i0], ui = im[i0];
      re[i1] = ur - tr; im[i1] = ui - ti;
      re[i0] = ur + tr; im[i0] = ui + ti;
    }
    __syncthreads();
  }
  if (inverse) {
    const float invn = 1.0f / (float)NFFT;
    for (int idx = t; idx < NFFT; idx += 256) { re[idx] *= invn; im[idx] *= invn; }
    __syncthreads();
  }
}

// ---------------------------------------------------------------------------
// RTF spectrum precompute: FK[c] = FFT( trunc_L( IFFT( FFT(B)/FFT([1,A]) ) ) )
// ---------------------------------------------------------------------------
__global__ __launch_bounds__(256) void rtf_fk_kernel(const float* __restrict__ A,
                                                     const float* __restrict__ Bp,
                                                     float* __restrict__ FK) {
  __shared__ float re[NFFT];
  __shared__ float im[NFFT];
  const int c = blockIdx.x;
  const int t = threadIdx.x;
  for (int idx = t; idx < NFFT; idx += 256) {
    float v = 0.f;
    if (idx == 0) v = 1.f;
    else if (idx <= H_ORD) v = A[c * H_ORD + (idx - 1)];
    re[idx] = v; im[idx] = 0.f;
  }
  __syncthreads();
  fft4096_lds(re, im, false);
  float far[16], fai[16];
#pragma unroll
  for (int j = 0; j < 16; ++j) { int idx = t + 256 * j; far[j] = re[idx]; fai[j] = im[idx]; }
  __syncthreads();
  for (int idx = t; idx < NFFT; idx += 256) {
    re[idx] = (idx < H_ORD) ? Bp[c * H_ORD + idx] : 0.f;
    im[idx] = 0.f;
  }
  __syncthreads();
  fft4096_lds(re, im, false);
#pragma unroll
  for (int j = 0; j < 16; ++j) {
    int idx = t + 256 * j;
    float br = re[idx], bi = im[idx];
    float invd = 1.0f / (far[j] * far[j] + fai[j] * fai[j]);
    re[idx] = (br * far[j] + bi * fai[j]) * invd;
    im[idx] = (bi * far[j] - br * fai[j]) * invd;
  }
  __syncthreads();
  fft4096_lds(re, im, true);
  for (int idx = t; idx < NFFT; idx += 256) {
    re[idx] = (idx < L_SEQ) ? re[idx] : 0.f;
    im[idx] = 0.f;
  }
  __syncthreads();
  fft4096_lds(re, im, false);
  for (int idx = t; idx < NFFT; idx += 256) {
    FK[((size_t)c * NFFT + idx) * 2 + 0] = re[idx];
    FK[((size_t)c * NFFT + idx) * 2 + 1] = im[idx];
  }
}

// ---------------------------------------------------------------------------
// FFT convolution: out[b,:,c] = irfft(rfft(pad(u[b,:,c])) * FK[c])[:L] + h0*u
// ---------------------------------------------------------------------------
__global__ __launch_bounds__(256) void fftconv_kernel(const float* __restrict__ u,
                                                      const float* __restrict__ FK,
                                                      const float* __restrict__ h0,
                                                      float* __restrict__ out) {
  __shared__ float re[NFFT];
  __shared__ float im[NFFT];
  const int c = blockIdx.x;
  const int b = blockIdx.y;
  const int t = threadIdx.x;
  const size_t base = ((size_t)b * L_SEQ) * C_CH + c;
  for (int idx = t; idx < NFFT; idx += 256) {
    re[idx] = (idx < L_SEQ) ? u[base + (size_t)idx * C_CH] : 0.f;
    im[idx] = 0.f;
  }
  __syncthreads();
  fft4096_lds(re, im, false);
  for (int idx = t; idx < NFFT; idx += 256) {
    float kr = FK[((size_t)c * NFFT + idx) * 2 + 0];
    float ki = FK[((size_t)c * NFFT + idx) * 2 + 1];
    float xr = re[idx], xi = im[idx];
    re[idx] = xr * kr - xi * ki;
    im[idx] = xr * ki + xi * kr;
  }
  __syncthreads();
  fft4096_lds(re, im, true);
  const float h0v = h0[0];
  for (int idx = t; idx < L_SEQ; idx += 256) {
    out[base + (size_t)idx * C_CH] = re[idx] + h0v * u[base + (size_t)idx * C_CH];
  }
}

// ---------------------------------------------------------------------------
// WMMA GEMM:  C[M,N] = A[M,K] * Bt^T[K,N] (+bias)
//   A  : M x K row-major f16
//   Bt : N x K row-major f16  (i.e. B transposed -> K contiguous per row)
// Block = 8 wave32 waves -> 64x32 C tile; double-buffered async-LDS staging:
// each thread issues one async b128 (A) + one async b64 (B) per K tile.
// BIAS: 0 none, 1 per-column bias[n], 2 per-row bias[m].
// ---------------------------------------------------------------------------
template <int BIAS>
__global__ __launch_bounds__(256) void gemm_wmma_kernel(
    const _Float16* __restrict__ A, const _Float16* __restrict__ Bt,
    float* __restrict__ Cm, const float* __restrict__ bias,
    int M, int N, int K,
    size_t strideA, size_t strideB, size_t strideC) {
  __shared__ __align__(16) _Float16 AshD[2][64 * 32]; // [m][k], stride 32
  __shared__ __align__(16) _Float16 BshD[2][32 * 32]; // [n][k], stride 32
  const int z = blockIdx.z;
  A  += (size_t)z * strideA;
  Bt += (size_t)z * strideB;
  Cm += (size_t)z * strideC;
  const int bm = blockIdx.x * 64;
  const int bn = blockIdx.y * 32;
  const int tid  = threadIdx.x;
  const int lane = tid & 31;
  const int wave = tid >> 5;         // 0..7
  const int wm = (wave & 3) * 16;
  const int wn = (wave >> 2) * 16;

  // per-thread staging chunks: A -> 16B, B -> 8B (LDS offsets are linear in tid)
  const unsigned ldsA0 = (unsigned)(size_t)&AshD[0][tid * 8];
  const unsigned ldsA1 = (unsigned)(size_t)&AshD[1][tid * 8];
  const unsigned ldsB0 = (unsigned)(size_t)&BshD[0][tid * 4];
  const unsigned ldsB1 = (unsigned)(size_t)&BshD[1][tid * 4];
  const _Float16* gA = A  + (size_t)(bm + (tid >> 2)) * K + (tid & 3) * 8;
  const _Float16* gB = Bt + (size_t)(bn + (tid >> 3)) * K + (tid & 7) * 4;

  const int nt = K >> 5;
  // prologue: stage tile 0 into buffer 0
  asm volatile("global_load_async_to_lds_b128 %0, %1, off"
               :: "v"(ldsA0), "v"((unsigned long long)(size_t)gA) : "memory");
  asm volatile("global_load_async_to_lds_b64 %0, %1, off"
               :: "v"(ldsB0), "v"((unsigned long long)(size_t)gB) : "memory");

  float8v acc = {};
  for (int kt = 0; kt < nt; ++kt) {
    const int cur = kt & 1;
    asm volatile("s_wait_asynccnt 0x0" ::: "memory");
    __syncthreads();
    if (kt + 1 < nt) {   // stage next tile into the other buffer (overlaps WMMA)
      gA += 32; gB += 32;
      __builtin_prefetch(gA, 0, 1);
      const unsigned la = cur ? ldsA0 : ldsA1;
      const unsigned lb = cur ? ldsB0 : ldsB1;
      asm volatile("global_load_async_to_lds_b128 %0, %1, off"
                   :: "v"(la), "v"((unsigned long long)(size_t)gA) : "memory");
      asm volatile("global_load_async_to_lds_b64 %0, %1, off"
                   :: "v"(lb), "v"((unsigned long long)(size_t)gB) : "memory");
    }
    const _Float16* Ab = AshD[cur];
    const _Float16* Bb = BshD[cur];

    // A fragment: lanes 0-15 row M=lane, halves 0..7=K0..7, 8..15=K16..23;
    // lanes 16-31 row M=lane-16, halves 0..7=K8..15, 8..15=K24..31.
    const int arow  = wm + (lane & 15);
    const int akoff = (lane < 16) ? 0 : 8;
    half8v alo = *(const half8v*)&Ab[arow * 32 + akoff];
    half8v ahi = *(const half8v*)&Ab[arow * 32 + 16 + akoff];
    half16v afrag = __builtin_shufflevector(alo, ahi,
        0, 1, 2, 3, 4, 5, 6, 7, 8, 9, 10, 11, 12, 13, 14, 15);

    // B fragment: lanes 0-15: N=lane, K=0..7,16..23; lanes 16-31: N=lane-16, K=8..15,24..31.
    const int brow  = wn + (lane & 15);
    const int bkoff = (lane < 16) ? 0 : 8;
    half8v blo = *(const half8v*)&Bb[brow * 32 + bkoff];
    half8v bhi = *(const half8v*)&Bb[brow * 32 + 16 + bkoff];
    half16v bfrag = __builtin_shufflevector(blo, bhi,
        0, 1, 2, 3, 4, 5, 6, 7, 8, 9, 10, 11, 12, 13, 14, 15);

    acc = __builtin_amdgcn_wmma_f32_16x16x32_f16(false, afrag, false, bfrag,
                                                 (short)0, acc, false, false);
  }

  // C layout: VGPR r -> lanes 0-15: (M=r, N=lane); lanes 16-31: (M=8+r, N=lane-16)
  const int col = bn + wn + (lane & 15);
  const int rbase = bm + wm + ((lane < 16) ? 0 : 8);
#pragma unroll
  for (int r = 0; r < 8; ++r) {
    int row = rbase + r;
    float v = acc[r];
    if (BIAS == 1)      v += bias[col];
    else if (BIAS == 2) v += bias[row];
    Cm[(size_t)row * N + col] = v;
  }
}

// ---------------------------------------------------------------------------
// Elementwise kernels
// ---------------------------------------------------------------------------
__global__ __launch_bounds__(256) void ew_cvt_f16(const float* __restrict__ s,
                                                  _Float16* __restrict__ d, size_t n) {
  size_t i = (size_t)blockIdx.x * 256 + threadIdx.x;
  if (i < n) d[i] = (_Float16)s[i];
}

// LDS-tiled transposing convert: in (rows x cols) f32 -> out (cols x rows) f16
// blockIdx.z = batch (strides rows*cols)
__global__ __launch_bounds__(256) void ew_cvtT_f16(const float* __restrict__ in,
                                                   _Float16* __restrict__ out,
                                                   int rows, int cols) {
  __shared__ float tile[32][33];
  const size_t bofs = (size_t)blockIdx.z * rows * cols;
  const int r0 = blockIdx.x * 32;
  const int c0 = blockIdx.y * 32;
  const int tx = threadIdx.x & 31;
  const int ty = threadIdx.x >> 5;   // 0..7
#pragma unroll
  for (int i = 0; i < 4; ++i) {
    int r = ty + i * 8;
    tile[r][tx] = in[bofs + (size_t)(r0 + r) * cols + (c0 + tx)];
  }
  __syncthreads();
#pragma unroll
  for (int i = 0; i < 4; ++i) {
    int c = ty + i * 8;
    out[bofs + (size_t)(c0 + c) * rows + (r0 + tx)] = (_Float16)tile[tx][c];
  }
}

__global__ __launch_bounds__(256) void ew_sub_scaled(const float* __restrict__ xe,
                                                     const float* __restrict__ feed,
                                                     const float* __restrict__ gamma,
                                                     float* __restrict__ u, size_t n) {
  size_t i = (size_t)blockIdx.x * 256 + threadIdx.x;
  if (i < n) u[i] = xe[i] - gamma[0] * feed[i];
}

__device__ __forceinline__ float gelu_erf(float v) {
  return 0.5f * v * (1.0f + erff(v * 0.70710678118654752440f));
}

__global__ __launch_bounds__(256) void ew_gelu(const float* __restrict__ s,
                                               float* __restrict__ d, size_t n) {
  size_t i = (size_t)blockIdx.x * 256 + threadIdx.x;
  if (i < n) d[i] = gelu_erf(s[i]);
}

__global__ __launch_bounds__(256) void ew_gelu_add(const float* __restrict__ s,
                                                   const float* __restrict__ add,
                                                   float* __restrict__ d, size_t n) {
  size_t i = (size_t)blockIdx.x * 256 + threadIdx.x;
  if (i < n) d[i] = gelu_erf(s[i]) + add[i];
}

__global__ __launch_bounds__(256) void ew_tanh(const float* __restrict__ s,
                                               float* __restrict__ d, size_t n) {
  size_t i = (size_t)blockIdx.x * 256 + threadIdx.x;
  if (i < n) d[i] = tanhf(s[i]);
}

__global__ __launch_bounds__(256) void ew_relax(float* __restrict__ feed,
                                                const float* __restrict__ fb, size_t n) {
  size_t i = (size_t)blockIdx.x * 256 + threadIdx.x;
  if (i < n) feed[i] = feed[i] + 0.7f * (fb[i] - feed[i]);
}

// ---------------------------------------------------------------------------
// Anderson acceleration helpers
// ---------------------------------------------------------------------------
__global__ __launch_bounds__(256) void and_dots_kernel(const float* __restrict__ hx,
                                                       const float* __restrict__ hf,
                                                       float* __restrict__ gtg,
                                                       float* __restrict__ gtf, int k) {
  const int b = blockIdx.x;
  const int pid = blockIdx.y;
  const int km1 = k - 1;
  int i, j; bool isGTF;
  if (pid < km1 * km1) { i = pid / km1; j = pid % km1; isGTF = false; }
  else { i = pid - km1 * km1; j = 0; isGTF = true; }
  const size_t off0 = (size_t)b * D_PER_B;
  float sum = 0.f;
  for (int d = threadIdx.x; d < D_PER_B; d += 256) {
    size_t off = off0 + d;
    float gi = (hf[(size_t)(i + 1) * TOT_ELEMS + off] - hx[(size_t)(i + 1) * TOT_ELEMS + off])
             - (hf[(size_t)i * TOT_ELEMS + off] - hx[(size_t)i * TOT_ELEMS + off]);
    float o;
    if (isGTF) {
      o = hf[(size_t)(k - 1) * TOT_ELEMS + off] - hx[(size_t)(k - 1) * TOT_ELEMS + off];
    } else {
      o = (hf[(size_t)(j + 1) * TOT_ELEMS + off] - hx[(size_t)(j + 1) * TOT_ELEMS + off])
        - (hf[(size_t)j * TOT_ELEMS + off] - hx[(size_t)j * TOT_ELEMS + off]);
    }
    sum += gi * o;
  }
  __shared__ float red[256];
  red[threadIdx.x] = sum;
  __syncthreads();
  for (int s = 128; s > 0; s >>= 1) {
    if (threadIdx.x < s) red[threadIdx.x] += red[threadIdx.x + s];
    __syncthreads();
  }
  if (threadIdx.x == 0) {
    if (isGTF) gtf[b * 4 + i] = red[0];
    else       gtg[b * 16 + i * 4 + j] = red[0];
  }
}

__global__ void and_solve_kernel(const float* __restrict__ gtg,
                                 const float* __restrict__ gtf,
                                 float* __restrict__ alpha, int km1) {
  int b = threadIdx.x;
  if (b >= B_SZ) return;
  float Am[4][5];
  for (int i = 0; i < km1; ++i) {
    for (int j = 0; j < km1; ++j)
      Am[i][j] = gtg[b * 16 + i * 4 + j] + ((i == j) ? AND_LAM : 0.f);
    Am[i][km1] = -gtf[b * 4 + i];
  }
  for (int p = 0; p < km1; ++p) {
    float inv = 1.0f / Am[p][p];
    for (int j = p; j <= km1; ++j) Am[p][j] *= inv;
    for (int i = 0; i < km1; ++i) {
      if (i == p) continue;
      float f = Am[i][p];
      for (int j = p; j <= km1; ++j) Am[i][j] -= f * Am[p][j];
    }
  }
  for (int i = 0; i < 4; ++i) alpha[b * 4 + i] = (i < km1) ? Am[i][km1] : 0.f;
}

__global__ __launch_bounds__(256) void and_combine_kernel(const float* __restrict__ fb,
                                                          const float* __restrict__ hf,
                                                          const float* __restrict__ alpha,
                                                          float* __restrict__ feed, int km1) {
  size_t idx = (size_t)blockIdx.x * 256 + threadIdx.x;
  if (idx >= TOT_ELEMS) return;
  int b = (int)(idx / D_PER_B);
  float asum = 0.f, acc = 0.f;
  for (int i = 0; i < km1; ++i) {
    float al = alpha[b * 4 + i];
    asum += al;
    acc  += al * hf[(size_t)i * TOT_ELEMS + idx];
  }
  feed[idx] = (1.0f - asum) * fb[idx] + acc;
}

// ---------------------------------------------------------------------------
// Host orchestration
// ---------------------------------------------------------------------------
extern "C" void kernel_launch(void* const* d_in, const int* in_sizes, int n_in,
                              void* d_out, int out_size, void* d_ws, size_t ws_size,
                              hipStream_t stream) {
  (void)in_sizes; (void)n_in; (void)out_size; (void)ws_size;
  const float* x        = (const float*)d_in[0];
  const float* gamma    = (const float*)d_in[1];
  const float* enc_w_i  = (const float*)d_in[2];
  const float* enc_b_i  = (const float*)d_in[3];
  const float* A_i      = (const float*)d_in[4];
  const float* B_i      = (const float*)d_in[5];
  const float* h0_i     = (const float*)d_in[6];
  const float* fc_w_i   = (const float*)d_in[7];
  const float* fc_b_i   = (const float*)d_in[8];
  const float* dec_w_i  = (const float*)d_in[9];
  const float* dec_b_i  = (const float*)d_in[10];
  const float* enc_w_o  = (const float*)d_in[11];
  const float* enc_b_o  = (const float*)d_in[12];
  const float* A_o      = (const float*)d_in[13];
  const float* B_o      = (const float*)d_in[14];
  const float* h0_o     = (const float*)d_in[15];
  const float* fc_w_o   = (const float*)d_in[16];
  const float* fc_b_o   = (const float*)d_in[17];
  const float* dec_w_o  = (const float*)d_in[18];
  const float* dec_b_o  = (const float*)d_in[19];
  const float* fc2_w    = (const float*)d_in[20];
  const float* fc2_b    = (const float*)d_in[21];
  const float* fc1_w    = (const float*)d_in[22];
  const float* fc1_b    = (const float*)d_in[23];

  char* base = (char*)d_ws;
  size_t off = 0;
  auto alloc = [&](size_t bytes) -> void* {
    void* p = base + off;
    off += (bytes + 255) & ~(size_t)255;
    return p;
  };
  const size_t TOTf = TOT_ELEMS * sizeof(float);
  _Float16* xh    = (_Float16*)alloc(TOT_ELEMS * 2);
  _Float16* actH  = (_Float16*)alloc(TOT_ELEMS * 2);
  _Float16* f2hT  = (_Float16*)alloc(TOT_ELEMS * 2);  // per-batch (C x L), transposed
  _Float16* wEncI = (_Float16*)alloc(C_CH * C_CH * 2);
  _Float16* wFcI  = (_Float16*)alloc(C_CH * C_CH * 2);
  _Float16* wDecI = (_Float16*)alloc(C_CH * C_CH * 2);
  _Float16* wEncO = (_Float16*)alloc(C_CH * C_CH * 2);
  _Float16* wFcO  = (_Float16*)alloc(C_CH * C_CH * 2);
  _Float16* wDecO = (_Float16*)alloc(C_CH * C_CH * 2);
  _Float16* wFc2  = (_Float16*)alloc(C_CH * C_CH * 2);
  _Float16* wFc1h = (_Float16*)alloc((size_t)L_SEQ * L_SEQ * 2);
  float* FKi   = (float*)alloc((size_t)C_CH * NFFT * 2 * sizeof(float));
  float* FKo   = (float*)alloc((size_t)C_CH * NFFT * 2 * sizeof(float));
  float* feed  = (float*)alloc(TOTf);
  float* fbv   = (float*)alloc(TOTf);
  float* xe    = (float*)alloc(TOTf);
  float* ubuf  = (float*)alloc(TOTf);
  float* sbuf  = (float*)alloc(TOTf);
  float* y1    = (float*)alloc(TOTf);
  float* tbuf  = (float*)alloc(TOTf);
  float* y2    = (float*)alloc(TOTf);
  float* yblk  = (float*)alloc(TOTf);
  float* y3    = (float*)alloc(TOTf);
  float* f2    = (float*)alloc(TOTf);
  float* fm    = (float*)alloc(TOTf);
  float* hx    = (float*)alloc(TOTf * N_ITER);
  float* hf    = (float*)alloc(TOTf * N_ITER);
  float* gtg   = (float*)alloc(B_SZ * 16 * sizeof(float));
  float* gtf   = (float*)alloc(B_SZ * 4 * sizeof(float));
  float* alpha = (float*)alloc(B_SZ * 4 * sizeof(float));

  const int EWG = (int)((TOT_ELEMS + 255) / 256);
  const int WWG = (C_CH * C_CH + 255) / 256;

  auto gemm = [&](const _Float16* A, const _Float16* Bt, float* Cm, const float* bias,
                  int M, int N, int K, int biasMode, int batches,
                  size_t sA, size_t sB, size_t sC) {
    dim3 grid(M / 64, N / 32, batches);
    if (biasMode == 1)
      gemm_wmma_kernel<1><<<grid, 256, 0, stream>>>(A, Bt, Cm, bias, M, N, K, sA, sB, sC);
    else if (biasMode == 2)
      gemm_wmma_kernel<2><<<grid, 256, 0, stream>>>(A, Bt, Cm, bias, M, N, K, sA, sB, sC);
    else
      gemm_wmma_kernel<0><<<grid, 256, 0, stream>>>(A, Bt, Cm, bias, M, N, K, sA, sB, sC);
  };

  // --- one-time precompute ---
  // Bt for channel GEMMs is W itself (N=out rows, K=in cols) -> plain convert.
  ew_cvt_f16<<<EWG, 256, 0, stream>>>(x, xh, TOT_ELEMS);
  ew_cvt_f16<<<WWG, 256, 0, stream>>>(enc_w_i, wEncI, (size_t)C_CH * C_CH);
  ew_cvt_f16<<<WWG, 256, 0, stream>>>(fc_w_i,  wFcI,  (size_t)C_CH * C_CH);
  ew_cvt_f16<<<WWG, 256, 0, stream>>>(dec_w_i, wDecI, (size_t)C_CH * C_CH);
  ew_cvt_f16<<<WWG, 256, 0, stream>>>(enc_w_o, wEncO, (size_t)C_CH * C_CH);
  ew_cvt_f16<<<WWG, 256, 0, stream>>>(fc_w_o,  wFcO,  (size_t)C_CH * C_CH);
  ew_cvt_f16<<<WWG, 256, 0, stream>>>(dec_w_o, wDecO, (size_t)C_CH * C_CH);
  ew_cvt_f16<<<WWG, 256, 0, stream>>>(fc2_w,   wFc2,  (size_t)C_CH * C_CH);
  ew_cvt_f16<<<(int)(((size_t)L_SEQ * L_SEQ + 255) / 256), 256, 0, stream>>>(
      fc1_w, wFc1h, (size_t)L_SEQ * L_SEQ);
  rtf_fk_kernel<<<C_CH, 256, 0, stream>>>(A_i, B_i, FKi);
  rtf_fk_kernel<<<C_CH, 256, 0, stream>>>(A_o, B_o, FKo);
  hipMemsetAsync(feed, 0, TOTf, stream);

  auto core_block = [&](const _Float16* inH, const float* feedPtr,
                        const _Float16* wEnc, const float* bEnc,
                        const _Float16* wFc, const float* bFc,
                        const _Float16* wDec, const float* bDec,
                        const float* FK, const float* h0, float* outBlk) {
    gemm(inH, wEnc, xe, bEnc, M_ROWS, C_CH, C_CH, 1, 1, 0, 0, 0);
    const float* uPtr = xe;
    if (feedPtr) {
      ew_sub_scaled<<<EWG, 256, 0, stream>>>(xe, feedPtr, gamma, ubuf, TOT_ELEMS);
      uPtr = ubuf;
    }
    fftconv_kernel<<<dim3(C_CH, B_SZ), 256, 0, stream>>>(uPtr, FK, h0, sbuf);
    ew_gelu<<<EWG, 256, 0, stream>>>(sbuf, y1, TOT_ELEMS);
    ew_cvt_f16<<<EWG, 256, 0, stream>>>(y1, actH, TOT_ELEMS);
    gemm(actH, wFc, tbuf, bFc, M_ROWS, C_CH, C_CH, 1, 1, 0, 0, 0);
    ew_gelu_add<<<EWG, 256, 0, stream>>>(tbuf, xe, y2, TOT_ELEMS);
    ew_cvt_f16<<<EWG, 256, 0, stream>>>(y2, actH, TOT_ELEMS);
    gemm(actH, wDec, outBlk, bDec, M_ROWS, C_CH, C_CH, 1, 1, 0, 0, 0);
  };

  auto model_pass = [&](const float* feedPtr, float* y3out, float* fbOut) {
    core_block(xh, feedPtr, wEncI, enc_b_i, wFcI, fc_b_i, wDecI, dec_b_i, FKi, h0_i, yblk);
    ew_cvt_f16<<<EWG, 256, 0, stream>>>(yblk, actH, TOT_ELEMS);
    core_block(actH, nullptr, wEncO, enc_b_o, wFcO, fc_b_o, wDecO, dec_b_o, FKo, h0_o, y3out);
    if (fbOut) {
      ew_cvt_f16<<<EWG, 256, 0, stream>>>(y3out, actH, TOT_ELEMS);
      gemm(actH, wFc2, f2, fc2_b, M_ROWS, C_CH, C_CH, 1, 1, 0, 0, 0);
      // transpose+convert f2 (B, L, C) -> f2hT (B, C, L) for the fc1 GEMM B operand
      ew_cvtT_f16<<<dim3(L_SEQ / 32, C_CH / 32, B_SZ), 256, 0, stream>>>(f2, f2hT, L_SEQ, C_CH);
      // per-batch: fm[b] (L x C) = fc1_w (L x L) @ f2[b] (L x C) + fc1_b[row]
      gemm(wFc1h, f2hT, fm, fc1_b, L_SEQ, C_CH, L_SEQ, 2, B_SZ,
           0, (size_t)D_PER_B, (size_t)D_PER_B);
      ew_tanh<<<EWG, 256, 0, stream>>>(fm, fbOut, TOT_ELEMS);
    }
  };

  // --- fixed-point iterations with Anderson acceleration ---
  for (int p = 0; p < N_ITER; ++p) {
    model_pass(feed, y3, fbv);
    hipMemcpyAsync(hx + (size_t)p * TOT_ELEMS, feed, TOTf, hipMemcpyDeviceToDevice, stream);
    hipMemcpyAsync(hf + (size_t)p * TOT_ELEMS, fbv,  TOTf, hipMemcpyDeviceToDevice, stream);
    const int k = p + 1;
    if (k < 2) {
      ew_relax<<<EWG, 256, 0, stream>>>(feed, fbv, TOT_ELEMS);
    } else {
      const int km1 = k - 1;
      and_dots_kernel<<<dim3(B_SZ, km1 * km1 + km1), 256, 0, stream>>>(hx, hf, gtg, gtf, k);
      and_solve_kernel<<<1, B_SZ, 0, stream>>>(gtg, gtf, alpha, km1);
      and_combine_kernel<<<EWG, 256, 0, stream>>>(fbv, hf, alpha, feed, km1);
    }
  }

  // final differentiable pass
  model_pass(feed, y3, nullptr);
  hipMemcpyAsync(d_out, y3, TOTf, hipMemcpyDeviceToDevice, stream);
}
